// weigaibianceluo_51015621542229
// MI455X (gfx1250) — matrix operand order repaired and checked by reading, not compiled, not run on hardware
//
#include <hip/hip_runtime.h>
#include <hip/hip_bf16.h>

typedef __attribute__((ext_vector_type(16))) _Float16 v16h;
typedef __attribute__((ext_vector_type(8)))  _Float16 v8h;
typedef __attribute__((ext_vector_type(8)))  float    v8f;

#define NUM_G    4096
#define NREAL    31
#define NNODE    32
#define EG       93
#define E_TOT    (NUM_G*EG)      // 380928
#define N_TOT    (NUM_G*NNODE)   // 131072
#define KDIM     128             // C_IN + C_EDGE
#define HD       256             // H*D
#define NH       4
#define DD       64
#define GROWS    125             // gather rows per graph (62 + 32 + 31)
#define BPAD     48              // padded per-column stride (f16) of a 32-K block
                                 // 96 bytes: keeps every 32B fragment chunk aligned

// A fragment: lane (half,mr) holds row mr,
// K = half*16+0..15 within the 32-K block -> two aligned v8h.
__device__ __forceinline__ v16h load_a_frag(const _Float16* rowBase, int kb, int half)
{
    v8h lo = *(const v8h*)(rowBase + kb + half * 16);
    v8h hi = *(const v8h*)(rowBase + kb + half * 16 + 8);
    return __builtin_shufflevector(lo, hi, 0,1,2,3,4,5,6,7,8,9,10,11,12,13,14,15);
}

// ---------------------------------------------------------------------------
// Kernel 1: h = [x[src] || edge_inform] @ W_lin_l   (E x 128) * (128 x 256)
// f16 WMMA 16x16x32, f32 accumulate. Block = 128 threads (4 waves).
// blockIdx.y selects which 128-column half of W this block computes.
// W staged in LDS in fragment-native swizzled layout:
//   sW[((k>>5)*128 + col)*BPAD + (k&31)]
// so a B fragment (lane col = wb+mr, K = kb+half*16+q) is one aligned v16h.
// ---------------------------------------------------------------------------
__global__ __launch_bounds__(128)
void k_gemm_h(const float* __restrict__ x,
              const int*   __restrict__ src,
              const float* __restrict__ einf,
              const float* __restrict__ W,
              float* __restrict__ h,
              int numTiles)
{
    __shared__ _Float16 sW[4 * 128 * BPAD];   // 48 KB swizzled W half (f16)
    __shared__ _Float16 sA[16 * KDIM];        // 4 KB A tile (row-major f16)

    const int tid     = threadIdx.x;
    const int colHalf = blockIdx.y;           // 0 -> cols [0,128), 1 -> [128,256)

    for (int i = tid; i < KDIM * 128; i += 128) {
        int k = i >> 7, c = i & 127;
        sW[((k >> 5) * 128 + c) * BPAD + (k & 31)] =
            (_Float16)W[k * HD + colHalf * 128 + c];
    }
    __syncthreads();

    const int wave = tid >> 5;
    const int lane = tid & 31;
    const int half = lane >> 4;
    const int mr   = lane & 15;
    const int wb   = wave * 32;               // local column base for this wave

    for (int tile = blockIdx.x; tile < numTiles; tile += gridDim.x) {
        const int e0 = tile * 16;

        // Stage A tile: rows = 16 edges, cols 0..63 = x[src], 64..127 = edge_inform
        for (int i = tid; i < 16 * KDIM; i += 128) {
            int r = i >> 7, c = i & 127;
            int e = e0 + r;
            float v = (c < 64) ? x[(long)src[e] * 64 + c]
                               : einf[(long)e * 64 + (c - 64)];
            sA[i] = (_Float16)v;
        }
        __syncthreads();

        const _Float16* aRow = sA + mr * KDIM;
        v8f acc0 = {}; v8f acc1 = {};
        #pragma unroll
        for (int kb = 0; kb < KDIM; kb += 32) {
            v16h af = load_a_frag(aRow, kb, half);
            const int kbb = kb >> 5;
            v16h bf0 = *(const v16h*)(sW + (kbb * 128 + wb + mr)      * BPAD + half * 16);
            v16h bf1 = *(const v16h*)(sW + (kbb * 128 + wb + 16 + mr) * BPAD + half * 16);
            acc0 = __builtin_amdgcn_wmma_f32_16x16x32_f16(false, af, false, bf0,
                                                          (short)0, acc0, false, false);
            acc1 = __builtin_amdgcn_wmma_f32_16x16x32_f16(false, af, false, bf1,
                                                          (short)0, acc1, false, false);
        }

        // D layout: VGPR r -> M = r + 8*half, N = mr
        const int cbase = colHalf * 128 + wb;
        #pragma unroll
        for (int r = 0; r < 8; ++r) {
            int m = r + half * 8;
            float* hp = h + (long)(e0 + m) * HD + cbase;
            hp[mr]      = acc0[r];
            hp[16 + mr] = acc1[r];
        }
        __syncthreads();
    }
}

// ---------------------------------------------------------------------------
// Kernel 2: per-graph attention softmax + aggregation -> out (N x 64).
// One block per graph. Graph structure is fixed: real node d receives ring
// edges 2*((d+30)%31) and 2d+1; virtual node 31 receives edges 62..92.
// Deterministic: no atomics, structured LDS reductions.
// ---------------------------------------------------------------------------
__global__ __launch_bounds__(256)
void k_node_out(const float* __restrict__ h,
                const float* __restrict__ att,
                const float* __restrict__ bias,
                float* __restrict__ out)
{
    __shared__ float sAtt[HD];        // att_l flat (4 heads x 64)
    __shared__ float sAl[EG * NH];    // leaky_relu(alpha) per (edge, head)
    __shared__ float sWt[EG * NH];    // softmax weight per (edge, head)

    const int g   = blockIdx.x;
    const int tid = threadIdx.x;
    const float* hg = h + (long)g * EG * HD;

    if (tid < HD) sAtt[tid] = att[tid];
    __syncthreads();

    // alpha = leaky_relu(sum_d h[e,hh,d] * att[hh,d])
    for (int i = tid; i < EG * NH; i += 256) {
        int e = i >> 2, hh = i & 3;
        const float* hp = hg + (long)e * HD + hh * DD;
        const float* ap = sAtt + hh * DD;
        float s = 0.f;
        #pragma unroll 8
        for (int d = 0; d < DD; ++d) s += hp[d] * ap[d];
        sAl[i] = (s > 0.f) ? s : 0.01f * s;
    }
    __syncthreads();

    // Softmax over in-edges of each destination node
    if (tid < 124) {                       // real nodes: (d, head)
        int d = tid >> 2, hh = tid & 3;
        int eA = 2 * ((d + 30) % 31), eB = 2 * d + 1;
        float aA = sAl[eA * 4 + hh], aB = sAl[eB * 4 + hh];
        float m  = fmaxf(aA, aB);
        float xA = expf(aA - m), xB = expf(aB - m);
        float inv = 1.f / (xA + xB + 1e-16f);
        sWt[eA * 4 + hh] = xA * inv;
        sWt[eB * 4 + hh] = xB * inv;
    } else if (tid < 128) {                // virtual node: one thread per head
        int hh = tid & 3;
        float m = -3.402823466e38f;
        for (int k = 0; k < NREAL; ++k) m = fmaxf(m, sAl[(62 + k) * 4 + hh]);
        float s = 0.f;
        for (int k = 0; k < NREAL; ++k) {
            float ex = expf(sAl[(62 + k) * 4 + hh] - m);
            sWt[(62 + k) * 4 + hh] = ex;
            s += ex;
        }
        float inv = 1.f / (s + 1e-16f);
        for (int k = 0; k < NREAL; ++k) sWt[(62 + k) * 4 + hh] *= inv;
    }
    __syncthreads();

    // out[node, c] = mean over heads of sum_e w[e,hh] * h[e,hh,c]  + bias[c]
    const int c = tid & 63, drow = tid >> 6;     // 4 nodes in flight
    const float bb = bias[c];
    for (int d = drow; d < NNODE; d += 4) {
        float acc = 0.f;
        if (d < NREAL) {
            int eA = 2 * ((d + 30) % 31), eB = 2 * d + 1;
            #pragma unroll
            for (int hh = 0; hh < NH; ++hh) {
                acc += sWt[eA * 4 + hh] * hg[(long)eA * HD + hh * DD + c];
                acc += sWt[eB * 4 + hh] * hg[(long)eB * HD + hh * DD + c];
            }
        } else {
            for (int k = 0; k < NREAL; ++k) {
                int e = 62 + k;
                #pragma unroll
                for (int hh = 0; hh < NH; ++hh)
                    acc += sWt[e * 4 + hh] * hg[(long)e * HD + hh * DD + c];
            }
        }
        out[((long)g * NNODE + d) * DD + c] = 0.25f * acc + bb;
    }
}

// ---------------------------------------------------------------------------
// Kernel 3: new_edge_inform = leaky_relu(gather @ W_ec + b_ec).
// gather rows are closed-form neighbor sums of edge_inform (ring structure):
//   r <  62: temp[r/2] = inf[2((j+30)%31)] + inf[2j+1] + inf[2((j+1)%31)+1]
//   62<=r<94: emean[d]; d<31: (inf[2((d+30)%31)] + inf[2d+1])/2; d==31: mean 62..92
//   r >= 94: emean[r-94] (d<31 formula)
// W_ec staged in swizzled fragment-native LDS layout; 16x16x32 WMMA + fused
// bias/leaky_relu epilogue.
// ---------------------------------------------------------------------------
__global__ __launch_bounds__(128)
void k_edge_out(const float* __restrict__ einf,
                const float* __restrict__ Wec,
                const float* __restrict__ bec,
                float* __restrict__ out2,
                int numTiles)
{
    __shared__ _Float16 sW[2 * 64 * BPAD];   // 12 KB swizzled W_ec (f16)
    __shared__ _Float16 sG[16 * 64];         // 2 KB gather tile (row-major)

    const int tid = threadIdx.x;
    for (int i = tid; i < 64 * 64; i += 128) {
        int k = i >> 6, c = i & 63;
        sW[((k >> 5) * 64 + c) * BPAD + (k & 31)] = (_Float16)Wec[i];
    }
    __syncthreads();

    const int wave = tid >> 5;
    const int lane = tid & 31;
    const int half = lane >> 4;
    const int mr   = lane & 15;
    const int nb   = wave * 16;              // each wave owns one 16-col tile

    for (int tile = blockIdx.x; tile < numTiles; tile += gridDim.x) {
        // Build 16 gather rows in LDS
        for (int i = tid; i < 16 * 64; i += 128) {
            int rl = i >> 6, c = i & 63;
            long row = (long)tile * 16 + rl;
            int g = (int)(row / GROWS), r = (int)(row % GROWS);
            const float* ig = einf + (long)g * EG * 64;
            float val;
            if (r < 62) {
                int j = r >> 1;
                val = ig[(2 * ((j + 30) % 31)) * 64 + c]
                    + ig[(2 * j + 1) * 64 + c]
                    + ig[(2 * ((j + 1) % 31) + 1) * 64 + c];
            } else if (r < 94) {
                int d = r - 62;
                if (d < NREAL) {
                    val = 0.5f * (ig[(2 * ((d + 30) % 31)) * 64 + c]
                                + ig[(2 * d + 1) * 64 + c]);
                } else {
                    float s = 0.f;
                    for (int k = 0; k < NREAL; ++k) s += ig[(62 + k) * 64 + c];
                    val = s * (1.0f / 31.0f);
                }
            } else {
                int d = r - 94;
                val = 0.5f * (ig[(2 * ((d + 30) % 31)) * 64 + c]
                            + ig[(2 * d + 1) * 64 + c]);
            }
            sG[i] = (_Float16)val;
        }
        __syncthreads();

        const _Float16* aRow = sG + mr * 64;
        v8f acc = {};
        #pragma unroll
        for (int kb = 0; kb < 64; kb += 32) {
            v16h af = load_a_frag(aRow, kb, half);
            v16h bf = *(const v16h*)(sW + ((kb >> 5) * 64 + nb + mr) * BPAD + half * 16);
            acc = __builtin_amdgcn_wmma_f32_16x16x32_f16(false, af, false, bf,
                                                         (short)0, acc, false, false);
        }

        // Epilogue: + b_ec, leaky_relu, store
        const int n = nb + mr;
        const float bb = bec[n];
        #pragma unroll
        for (int r = 0; r < 8; ++r) {
            int m = r + half * 8;
            long row = (long)tile * 16 + m;
            float v = acc[r] + bb;
            out2[row * 64 + n] = (v > 0.f) ? v : 0.01f * v;
        }
        __syncthreads();
    }
}

// ---------------------------------------------------------------------------
extern "C" void kernel_launch(void* const* d_in, const int* in_sizes, int n_in,
                              void* d_out, int out_size, void* d_ws, size_t ws_size,
                              hipStream_t stream)
{
    const float* x    = (const float*)d_in[0];
    const int*   ei   = (const int*)  d_in[1];   // (2, E) row-major
    const float* einf = (const float*)d_in[2];
    const float* Wlin = (const float*)d_in[5];
    const float* att  = (const float*)d_in[6];
    const float* bias = (const float*)d_in[7];
    const float* Wec  = (const float*)d_in[8];
    const float* bec  = (const float*)d_in[9];
    (void)in_sizes; (void)n_in; (void)out_size; (void)ws_size;

    float* out1 = (float*)d_out;                 // (N, 64)
    float* out2 = out1 + (long)N_TOT * DD;       // (G*125, 64)
    float* h    = (float*)d_ws;                  // (E, 256) f32 scratch
    const int* src = ei;                         // edge_index[0][:]

    const int tilesH = E_TOT / 16;               // 23808, exact
    dim3 gridH(2048, 2);
    k_gemm_h<<<gridH, 128, 0, stream>>>(x, src, einf, Wlin, h, tilesH);

    k_node_out<<<NUM_G, 256, 0, stream>>>(h, att, bias, out1);

    const int tilesE = (NUM_G * GROWS) / 16;     // 32000, exact
    k_edge_out<<<2048, 128, 0, stream>>>(einf, Wec, bec, out2, tilesE);
}
